// GraphQSat_40278203302105
// MI455X (gfx1250) — matrix operand order repaired and checked by reading, not compiled, not run on hardware
//
#include <hip/hip_runtime.h>

// ---------------------------------------------------------------------------
// GraphQSat forward for MI455X (gfx1250, wave32).
// Edge/Node MLPs run on V_WMMA_F32_16X16X32_BF16 (f32 accumulate).
// ---------------------------------------------------------------------------

typedef unsigned short ush;
typedef __attribute__((ext_vector_type(16))) __bf16 v16bf;
typedef __attribute__((ext_vector_type(8)))  float  v8f;
typedef __attribute__((ext_vector_type(4)))  unsigned int u32x4;

#define N_NODES  50000
#define N_EDGES  400000
#define N_EDGES2 800000
#define N_GRAPH  64

// ----- helpers -------------------------------------------------------------

__device__ __forceinline__ ush f2bf(float f) {           // round-to-nearest-even
  unsigned int u = __float_as_uint(f);
  u += 0x7FFFu + ((u >> 16) & 1u);
  return (ush)(u >> 16);
}
__device__ __forceinline__ float bf2f(ush h) {
  return __uint_as_float(((unsigned int)h) << 16);
}
__device__ __forceinline__ void atomAddF(float* p, float v) {
  __hip_atomic_fetch_add(p, v, __ATOMIC_RELAXED, __HIP_MEMORY_SCOPE_AGENT);
}
__device__ __forceinline__ v16bf load_frag16(const ush* p0, const ush* p1) {
  union { v16bf v; u32x4 q[2]; } u;
  u.q[0] = *reinterpret_cast<const u32x4*>(p0);
  u.q[1] = *reinterpret_cast<const u32x4*>(p1);
  return u.v;
}
__device__ __forceinline__ v8f wmma_bf16(v16bf a, v16bf b, v8f c) {
  return __builtin_amdgcn_wmma_f32_16x16x32_bf16(false, a, false, b, (short)0, c,
                                                 false, false);
}

// ----- weight transpose+convert: src[K][N] f32 -> dst[N][K] bf16 -----------

__global__ __launch_bounds__(256) void convT_kernel(const float* __restrict__ src,
                                                    ush* __restrict__ dst,
                                                    int K, int Nn) {
  int t = blockIdx.x * blockDim.x + threadIdx.x;
  if (t >= K * Nn) return;
  int nn = t / K, k = t - nn * K;
  dst[(size_t)nn * K + k] = f2bf(src[(size_t)k * Nn + nn]);
}

// ----- f32 -> bf16 elementwise --------------------------------------------

__global__ __launch_bounds__(256) void f32_to_bf16_kernel(const float* __restrict__ src,
                                                          ush* __restrict__ dst, int n) {
  int t = blockIdx.x * blockDim.x + threadIdx.x;
  if (t < n) dst[t] = f2bf(src[t]);
}

// ----- encoders ------------------------------------------------------------

__global__ __launch_bounds__(256) void encode_nodes_kernel(
    const float* __restrict__ x, const int* __restrict__ batch,
    const float* __restrict__ w, const float* __restrict__ b,
    const float* __restrict__ lng, const float* __restrict__ lnb,
    ush* __restrict__ x_enc, float* __restrict__ vw, float* __restrict__ vcnt,
    float* __restrict__ mask_out, int n) {
  int i = blockIdx.x * blockDim.x + threadIdx.x;
  if (i >= n) return;
  float x0 = x[2 * i], x1 = x[2 * i + 1];
  float a[32], s = 0.f, sq = 0.f;
#pragma unroll
  for (int j = 0; j < 32; ++j) {
    float v = fmaxf(x0 * w[j] + x1 * w[32 + j] + b[j], 0.f);
    a[j] = v; s += v; sq += v * v;
  }
  float mean = s * (1.f / 32.f);
  float var = sq * (1.f / 32.f) - mean * mean;
  float rs = rsqrtf(var + 1e-5f);
#pragma unroll
  for (int j = 0; j < 32; ++j)
    x_enc[(size_t)i * 32 + j] = f2bf((a[j] - mean) * rs * lng[j] + lnb[j]);
  float vv = (x0 > x1) ? 1.f : 0.f;
  vw[i] = vv;
  mask_out[i] = vv;
  atomAddF(&vcnt[batch[i]], vv);
}

__global__ __launch_bounds__(256) void encode_edges_kernel(
    const float* __restrict__ eattr, const int* __restrict__ ei0,
    const int* __restrict__ ei1, const int* __restrict__ batch,
    const float* __restrict__ w, const float* __restrict__ b,
    const float* __restrict__ lng, const float* __restrict__ lnb,
    ush* __restrict__ e_enc, float* __restrict__ ecnt, int E2, int E) {
  int e = blockIdx.x * blockDim.x + threadIdx.x;
  if (e >= E2) return;
  int r = (e < E) ? ei0[e] : ei1[e - E];
  atomAddF(&ecnt[batch[r]], 1.f);
  if (e >= E) return;                       // e_enc rows are shared between e and e+E
  float x0 = eattr[2 * e], x1 = eattr[2 * e + 1];
  float a[32], s = 0.f, sq = 0.f;
#pragma unroll
  for (int j = 0; j < 32; ++j) {
    float v = fmaxf(x0 * w[j] + x1 * w[32 + j] + b[j], 0.f);
    a[j] = v; s += v; sq += v * v;
  }
  float mean = s * (1.f / 32.f);
  float var = sq * (1.f / 32.f) - mean * mean;
  float rs = rsqrtf(var + 1e-5f);
#pragma unroll
  for (int j = 0; j < 32; ++j)
    e_enc[(size_t)e * 32 + j] = f2bf((a[j] - mean) * rs * lng[j] + lnb[j]);
}

// u0 == 0 so u_enc is one shared 32-vector: LN(relu(bias))
__global__ void encode_glob_kernel(const float* __restrict__ b,
                                   const float* __restrict__ lng,
                                   const float* __restrict__ lnb,
                                   ush* __restrict__ u_enc) {
  if (threadIdx.x != 0 || blockIdx.x != 0) return;
  float a[32], s = 0.f, sq = 0.f;
  for (int j = 0; j < 32; ++j) {
    float v = fmaxf(b[j], 0.f);
    a[j] = v; s += v; sq += v * v;
  }
  float mean = s * (1.f / 32.f);
  float var = sq * (1.f / 32.f) - mean * mean;
  float rs = rsqrtf(var + 1e-5f);
  for (int j = 0; j < 32; ++j)
    u_enc[j] = f2bf((a[j] - mean) * rs * lng[j] + lnb[j]);
}

// ----- edge MLP: WMMA, K=352 (11 chunks), N=64 (4 tiles) -------------------
// ein = [e_enc(32)|e_core(64)|x_enc[r](32)|x_core[r](64)|x_enc[c](32)|
//        x_core[c](64)|u_enc(32)|u_core[gb](32)]  -- all boundaries %32==0.

__global__ __launch_bounds__(128) void edge_mlp_kernel(
    const int* __restrict__ ei0, const int* __restrict__ ei1,
    const int* __restrict__ batch,
    const ush* __restrict__ e_enc, const ush* __restrict__ x_enc,
    const ush* __restrict__ u_enc, const ush* __restrict__ e_core_in,
    const ush* __restrict__ x_core_in, const ush* __restrict__ u_core_in,
    const ush* __restrict__ W1T, const float* __restrict__ b1,
    const ush* __restrict__ W2T, const float* __restrict__ b2,
    const float* __restrict__ lng, const float* __restrict__ lnb,
    ush* __restrict__ e_core_out, float* __restrict__ agg,
    float* __restrict__ e_sum, int E2, int E) {
  __shared__ ush   hTile[4][16 * 64];
  __shared__ float lnTile[4][16 * 64];
  const int lane = threadIdx.x & 31;
  const int wv = threadIdx.x >> 5;
  const int base = (blockIdx.x * 4 + wv) * 16;
  const bool active = base < E2;
  const int mrow = lane & 15;
  const int em = active ? (base + mrow) : 0;
  int r, c;
  if (em < E) { r = ei0[em]; c = ei1[em]; }
  else        { r = ei1[em - E]; c = ei0[em - E]; }
  const int gb = batch[r];
  const int ee = (em < E) ? em : (em - E);
  const int sel  = (lane < 16) ? 0 : 8;    // A-fragment K select
  const int bsel = (lane < 16) ? 0 : 8;    // B-fragment K select

  const ush* srcs[11];
  srcs[0]  = e_enc + (size_t)ee * 32;
  srcs[1]  = e_core_in + (size_t)em * 64;
  srcs[2]  = srcs[1] + 32;
  srcs[3]  = x_enc + (size_t)r * 32;
  srcs[4]  = x_core_in + (size_t)r * 64;
  srcs[5]  = srcs[4] + 32;
  srcs[6]  = x_enc + (size_t)c * 32;
  srcs[7]  = x_core_in + (size_t)c * 64;
  srcs[8]  = srcs[7] + 32;
  srcs[9]  = u_enc;
  srcs[10] = u_core_in + (size_t)gb * 32;

  v8f acc[4];
#pragma unroll
  for (int nt = 0; nt < 4; ++nt) {
    float bv = b1[nt * 16 + mrow];
#pragma unroll
    for (int i = 0; i < 8; ++i) acc[nt][i] = bv;
  }
#pragma unroll
  for (int kc = 0; kc < 11; ++kc) {
    const ush* ap = srcs[kc] + sel;
    v16bf a = load_frag16(ap, ap + 16);
#pragma unroll
    for (int nt = 0; nt < 4; ++nt) {
      const ush* bp = W1T + (size_t)(nt * 16 + mrow) * 352 + kc * 32 + bsel;
      v16bf b = load_frag16(bp, bp + 16);
      acc[nt] = wmma_bf16(a, b, acc[nt]);
    }
  }
  // ReLU -> bf16 h tile in LDS
#pragma unroll
  for (int nt = 0; nt < 4; ++nt) {
    int nn = nt * 16 + mrow;
#pragma unroll
    for (int i = 0; i < 8; ++i) {
      int mm = (lane < 16) ? i : (i + 8);
      hTile[wv][mm * 64 + nn] = f2bf(fmaxf(acc[nt][i], 0.f));
    }
  }
  __syncthreads();
  v8f a2[4];
#pragma unroll
  for (int nt = 0; nt < 4; ++nt) {
    float bv = b2[nt * 16 + mrow];
#pragma unroll
    for (int i = 0; i < 8; ++i) a2[nt][i] = bv;
  }
#pragma unroll
  for (int c2 = 0; c2 < 2; ++c2) {
    const ush* hp = &hTile[wv][mrow * 64 + c2 * 32 + sel];
    v16bf a = load_frag16(hp, hp + 16);
#pragma unroll
    for (int nt = 0; nt < 4; ++nt) {
      const ush* bp = W2T + (size_t)(nt * 16 + mrow) * 64 + c2 * 32 + bsel;
      v16bf b = load_frag16(bp, bp + 16);
      a2[nt] = wmma_bf16(a, b, a2[nt]);
    }
  }
#pragma unroll
  for (int nt = 0; nt < 4; ++nt) {
    int nn = nt * 16 + mrow;
#pragma unroll
    for (int i = 0; i < 8; ++i) {
      int mm = (lane < 16) ? i : (i + 8);
      lnTile[wv][mm * 64 + nn] = a2[nt][i];
    }
  }
  __syncthreads();
  // LayerNorm(64) + store + segment-sum atomics. lane & its pair handle row mrow.
  const int cb = (lane < 16) ? 0 : 32;
  const float* rowp = &lnTile[wv][mrow * 64 + cb];
  float s = 0.f, sq = 0.f;
#pragma unroll
  for (int j = 0; j < 32; ++j) { float v = rowp[j]; s += v; sq += v * v; }
  s  += __shfl_xor(s, 16, 32);
  sq += __shfl_xor(sq, 16, 32);
  float mean = s * (1.f / 64.f);
  float var = sq * (1.f / 64.f) - mean * mean;
  float rs = rsqrtf(var + 1e-5f);
  if (active) {
    float* aggp = agg + (size_t)c * 64;
    float* esp  = e_sum + (size_t)gb * 64;
    ush*   outp = e_core_out + (size_t)em * 64;
    for (int j = 0; j < 32; ++j) {
      int jj = cb + j;
      float v = (rowp[j] - mean) * rs * lng[jj] + lnb[jj];
      outp[jj] = f2bf(v);
      atomAddF(aggp + jj, v);
      atomAddF(esp + jj, v);
    }
  }
}

// ----- node MLP: WMMA, K=224 (7 chunks), N=64 ------------------------------
// nin = [x_enc(32)|x_core(64)|agg(64)|u_enc(32)|u_core[gb](32)]

__global__ __launch_bounds__(128) void node_mlp_kernel(
    const int* __restrict__ batch, const ush* __restrict__ x_enc,
    const ush* __restrict__ u_enc, const ush* __restrict__ x_core_in,
    const ush* __restrict__ agg_bf, const ush* __restrict__ u_core_in,
    const ush* __restrict__ W1T, const float* __restrict__ b1,
    const ush* __restrict__ W2T, const float* __restrict__ b2,
    const float* __restrict__ lng, const float* __restrict__ lnb,
    const float* __restrict__ vw, float* __restrict__ x_core_f32,
    ush* __restrict__ x_core_out, float* __restrict__ n_sum, int n) {
  __shared__ ush   hTile[4][16 * 64];
  __shared__ float lnTile[4][16 * 64];
  const int lane = threadIdx.x & 31;
  const int wv = threadIdx.x >> 5;
  const int base = (blockIdx.x * 4 + wv) * 16;
  const bool active = base < n;
  const int mrow = lane & 15;
  const int nd = active ? (base + mrow) : 0;
  const int gb = batch[nd];
  const int sel  = (lane < 16) ? 0 : 8;
  const int bsel = (lane < 16) ? 0 : 8;

  const ush* srcs[7];
  srcs[0] = x_enc + (size_t)nd * 32;
  srcs[1] = x_core_in + (size_t)nd * 64;
  srcs[2] = srcs[1] + 32;
  srcs[3] = agg_bf + (size_t)nd * 64;
  srcs[4] = srcs[3] + 32;
  srcs[5] = u_enc;
  srcs[6] = u_core_in + (size_t)gb * 32;

  v8f acc[4];
#pragma unroll
  for (int nt = 0; nt < 4; ++nt) {
    float bv = b1[nt * 16 + mrow];
#pragma unroll
    for (int i = 0; i < 8; ++i) acc[nt][i] = bv;
  }
#pragma unroll
  for (int kc = 0; kc < 7; ++kc) {
    const ush* ap = srcs[kc] + sel;
    v16bf a = load_frag16(ap, ap + 16);
#pragma unroll
    for (int nt = 0; nt < 4; ++nt) {
      const ush* bp = W1T + (size_t)(nt * 16 + mrow) * 224 + kc * 32 + bsel;
      v16bf b = load_frag16(bp, bp + 16);
      acc[nt] = wmma_bf16(a, b, acc[nt]);
    }
  }
#pragma unroll
  for (int nt = 0; nt < 4; ++nt) {
    int nn = nt * 16 + mrow;
#pragma unroll
    for (int i = 0; i < 8; ++i) {
      int mm = (lane < 16) ? i : (i + 8);
      hTile[wv][mm * 64 + nn] = f2bf(fmaxf(acc[nt][i], 0.f));
    }
  }
  __syncthreads();
  v8f a2[4];
#pragma unroll
  for (int nt = 0; nt < 4; ++nt) {
    float bv = b2[nt * 16 + mrow];
#pragma unroll
    for (int i = 0; i < 8; ++i) a2[nt][i] = bv;
  }
#pragma unroll
  for (int c2 = 0; c2 < 2; ++c2) {
    const ush* hp = &hTile[wv][mrow * 64 + c2 * 32 + sel];
    v16bf a = load_frag16(hp, hp + 16);
#pragma unroll
    for (int nt = 0; nt < 4; ++nt) {
      const ush* bp = W2T + (size_t)(nt * 16 + mrow) * 64 + c2 * 32 + bsel;
      v16bf b = load_frag16(bp, bp + 16);
      a2[nt] = wmma_bf16(a, b, a2[nt]);
    }
  }
#pragma unroll
  for (int nt = 0; nt < 4; ++nt) {
    int nn = nt * 16 + mrow;
#pragma unroll
    for (int i = 0; i < 8; ++i) {
      int mm = (lane < 16) ? i : (i + 8);
      lnTile[wv][mm * 64 + nn] = a2[nt][i];
    }
  }
  __syncthreads();
  const int cb = (lane < 16) ? 0 : 32;
  const float* rowp = &lnTile[wv][mrow * 64 + cb];
  float s = 0.f, sq = 0.f;
#pragma unroll
  for (int j = 0; j < 32; ++j) { float v = rowp[j]; s += v; sq += v * v; }
  s  += __shfl_xor(s, 16, 32);
  sq += __shfl_xor(sq, 16, 32);
  float mean = s * (1.f / 64.f);
  float var = sq * (1.f / 64.f) - mean * mean;
  float rs = rsqrtf(var + 1e-5f);
  if (active) {
    float vvw = vw[nd];
    float* nsp = n_sum + (size_t)gb * 64;
    for (int j = 0; j < 32; ++j) {
      int jj = cb + j;
      float v = (rowp[j] - mean) * rs * lng[jj] + lnb[jj];
      x_core_f32[(size_t)nd * 64 + jj] = v;
      x_core_out[(size_t)nd * 64 + jj] = f2bf(v);
      atomAddF(nsp + jj, v * vvw);
    }
  }
}

// ----- global MLP: G=64 rows, tiny -> plain VALU ---------------------------

__global__ __launch_bounds__(64) void glob_step_kernel(
    const ush* __restrict__ u_enc, const ush* __restrict__ u_core_in,
    const float* __restrict__ e_sum, const float* __restrict__ n_sum,
    const float* __restrict__ ecnt, const float* __restrict__ vcnt,
    const float* __restrict__ w1, const float* __restrict__ b1,
    const float* __restrict__ w2, const float* __restrict__ b2,
    const float* __restrict__ lng, const float* __restrict__ lnb,
    ush* __restrict__ u_core_out) {
  int g = threadIdx.x;
  if (g >= N_GRAPH) return;
  float gin[192];
  for (int j = 0; j < 32; ++j) gin[j] = bf2f(u_enc[j]);
  for (int j = 0; j < 32; ++j) gin[32 + j] = bf2f(u_core_in[g * 32 + j]);
  float ec = fmaxf(ecnt[g], 1.f), vc = fmaxf(vcnt[g], 1.f);
  for (int j = 0; j < 64; ++j) gin[64 + j] = e_sum[(size_t)g * 64 + j] / ec;
  for (int j = 0; j < 64; ++j) gin[128 + j] = n_sum[(size_t)g * 64 + j] / vc;
  float h[64];
  for (int nn = 0; nn < 64; ++nn) {
    float s = b1[nn];
    for (int k = 0; k < 192; ++k) s += gin[k] * w1[(size_t)k * 64 + nn];
    h[nn] = fmaxf(s, 0.f);
  }
  float o[32], s = 0.f, sq = 0.f;
  for (int nn = 0; nn < 32; ++nn) {
    float v = b2[nn];
    for (int k = 0; k < 64; ++k) v += h[k] * w2[(size_t)k * 32 + nn];
    o[nn] = v; s += v; sq += v * v;
  }
  float mean = s * (1.f / 32.f);
  float var = sq * (1.f / 32.f) - mean * mean;
  float rs = rsqrtf(var + 1e-5f);
  for (int nn = 0; nn < 32; ++nn)
    u_core_out[g * 32 + nn] = f2bf((o[nn] - mean) * rs * lng[nn] + lnb[nn]);
}

// ----- decoder: LN(relu(x_core @ Wd + bd)) @ Wq + bq -----------------------

__global__ __launch_bounds__(256) void decode_kernel(
    const float* __restrict__ x_core, const float* __restrict__ wd,
    const float* __restrict__ bd, const float* __restrict__ lng,
    const float* __restrict__ lnb, const float* __restrict__ wq,
    const float* __restrict__ bq, float* __restrict__ out, int n) {
  int i = blockIdx.x * blockDim.x + threadIdx.x;
  if (i >= n) return;
  float xr[64];
#pragma unroll
  for (int k = 0; k < 64; ++k) xr[k] = x_core[(size_t)i * 64 + k];
  float a[32], s = 0.f, sq = 0.f;
  for (int j = 0; j < 32; ++j) {
    float v = bd[j];
    for (int k = 0; k < 64; ++k) v += xr[k] * wd[(size_t)k * 32 + j];
    v = fmaxf(v, 0.f);
    a[j] = v; s += v; sq += v * v;
  }
  float mean = s * (1.f / 32.f);
  float var = sq * (1.f / 32.f) - mean * mean;
  float rs = rsqrtf(var + 1e-5f);
  float q0 = bq[0], q1 = bq[1];
  for (int j = 0; j < 32; ++j) {
    float hj = (a[j] - mean) * rs * lng[j] + lnb[j];
    q0 += hj * wq[2 * j];
    q1 += hj * wq[2 * j + 1];
  }
  out[2 * i] = q0;
  out[2 * i + 1] = q1;
}

// ---------------------------------------------------------------------------

extern "C" void kernel_launch(void* const* d_in, const int* in_sizes, int n_in,
                              void* d_out, int out_size, void* d_ws, size_t ws_size,
                              hipStream_t stream) {
  (void)in_sizes; (void)n_in; (void)out_size; (void)ws_size;
  const int N = N_NODES, E = N_EDGES, E2 = N_EDGES2, G = N_GRAPH;

  const float* x     = (const float*)d_in[0];
  const int*   ei0   = (const int*)d_in[1];
  const int*   ei1   = ei0 + E;
  const float* eattr = (const float*)d_in[2];
  const int*   batch = (const int*)d_in[3];
  // params: JAX pytree leaves, dict keys alphabetical, leaves (b,w)/(b,g) sorted
  const float* ee_lin_b = (const float*)d_in[4];
  const float* ee_lin_w = (const float*)d_in[5];
  const float* ee_ln_b  = (const float*)d_in[6];
  const float* ee_ln_g  = (const float*)d_in[7];
  const float* em_ln_b  = (const float*)d_in[8];
  const float* em_ln_g  = (const float*)d_in[9];
  const float* em_l1_b  = (const float*)d_in[10];
  const float* em_l1_w  = (const float*)d_in[11];
  const float* em_l2_b  = (const float*)d_in[12];
  const float* em_l2_w  = (const float*)d_in[13];
  const float* ge_lin_b = (const float*)d_in[14];
  const float* ge_ln_b  = (const float*)d_in[16];
  const float* ge_ln_g  = (const float*)d_in[17];
  const float* gm_ln_b  = (const float*)d_in[18];
  const float* gm_ln_g  = (const float*)d_in[19];
  const float* gm_l1_b  = (const float*)d_in[20];
  const float* gm_l1_w  = (const float*)d_in[21];
  const float* gm_l2_b  = (const float*)d_in[22];
  const float* gm_l2_w  = (const float*)d_in[23];
  const float* nd_lin_b = (const float*)d_in[24];
  const float* nd_lin_w = (const float*)d_in[25];
  const float* nd_ln_b  = (const float*)d_in[26];
  const float* nd_ln_g  = (const float*)d_in[27];
  const float* ne_lin_b = (const float*)d_in[28];
  const float* ne_lin_w = (const float*)d_in[29];
  const float* ne_ln_b  = (const float*)d_in[30];
  const float* ne_ln_g  = (const float*)d_in[31];
  const float* nm_ln_b  = (const float*)d_in[32];
  const float* nm_ln_g  = (const float*)d_in[33];
  const float* nm_l1_b  = (const float*)d_in[34];
  const float* nm_l1_w  = (const float*)d_in[35];
  const float* nm_l2_b  = (const float*)d_in[36];
  const float* nm_l2_w  = (const float*)d_in[37];
  const float* q_b      = (const float*)d_in[38];
  const float* q_w      = (const float*)d_in[39];

  // workspace carve-up (256B aligned)
  char* p = (char*)d_ws;
  auto alloc = [&](size_t bytes) -> char* {
    char* r = p;
    p += (bytes + 255) & ~(size_t)255;
    return r;
  };
  ush* e_coreA   = (ush*)alloc((size_t)E2 * 64 * 2);
  ush* e_coreB   = (ush*)alloc((size_t)E2 * 64 * 2);
  ush* e_enc_bf  = (ush*)alloc((size_t)E * 32 * 2);
  ush* x_enc_bf  = (ush*)alloc((size_t)N * 32 * 2);
  ush* x_coreA   = (ush*)alloc((size_t)N * 64 * 2);
  ush* x_coreB   = (ush*)alloc((size_t)N * 64 * 2);
  float* x_core_f32 = (float*)alloc((size_t)N * 64 * 4);
  float* agg     = (float*)alloc((size_t)N * 64 * 4);
  ush* agg_bf    = (ush*)alloc((size_t)N * 64 * 2);
  float* vw      = (float*)alloc((size_t)N * 4);
  ush* u_enc_bf  = (ush*)alloc(32 * 2);
  ush* u_coreA   = (ush*)alloc((size_t)G * 32 * 2);
  ush* u_coreB   = (ush*)alloc((size_t)G * 32 * 2);
  float* e_sum   = (float*)alloc((size_t)G * 64 * 4);
  float* n_sum   = (float*)alloc((size_t)G * 64 * 4);
  float* ecnt    = (float*)alloc((size_t)G * 4);
  float* vcnt    = (float*)alloc((size_t)G * 4);
  ush* W1Te      = (ush*)alloc((size_t)352 * 64 * 2);
  ush* W2Te      = (ush*)alloc((size_t)64 * 64 * 2);
  ush* W1Tn      = (ush*)alloc((size_t)224 * 64 * 2);
  ush* W2Tn      = (ush*)alloc((size_t)64 * 64 * 2);

  float* out_q    = (float*)d_out;
  float* out_mask = (float*)d_out + 2 * (size_t)N;

  // zero-init ping inputs (latents start as zeros) and counts
  hipMemsetAsync(e_coreA, 0, (size_t)E2 * 64 * 2, stream);
  hipMemsetAsync(x_coreA, 0, (size_t)N * 64 * 2, stream);
  hipMemsetAsync(u_coreA, 0, (size_t)G * 32 * 2, stream);
  hipMemsetAsync(ecnt, 0, (size_t)G * 4, stream);
  hipMemsetAsync(vcnt, 0, (size_t)G * 4, stream);

  // transpose + bf16-convert MLP weights
  convT_kernel<<<(352 * 64 + 255) / 256, 256, 0, stream>>>(em_l1_w, W1Te, 352, 64);
  convT_kernel<<<(64 * 64 + 255) / 256, 256, 0, stream>>>(em_l2_w, W2Te, 64, 64);
  convT_kernel<<<(224 * 64 + 255) / 256, 256, 0, stream>>>(nm_l1_w, W1Tn, 224, 64);
  convT_kernel<<<(64 * 64 + 255) / 256, 256, 0, stream>>>(nm_l2_w, W2Tn, 64, 64);

  // encoders
  encode_glob_kernel<<<1, 32, 0, stream>>>(ge_lin_b, ge_ln_g, ge_ln_b, u_enc_bf);
  encode_nodes_kernel<<<(N + 255) / 256, 256, 0, stream>>>(
      x, batch, ne_lin_w, ne_lin_b, ne_ln_g, ne_ln_b, x_enc_bf, vw, vcnt,
      out_mask, N);
  encode_edges_kernel<<<(E2 + 255) / 256, 256, 0, stream>>>(
      eattr, ei0, ei1, batch, ee_lin_w, ee_lin_b, ee_ln_g, ee_ln_b, e_enc_bf,
      ecnt, E2, E);

  const int edgeBlocks = E2 / 64;                 // 12500, exact
  const int nodeBlocks = (N / 16 + 3) / 4;        // 782

  for (int step = 0; step < 4; ++step) {
    const ush* e_in = (step & 1) ? e_coreB : e_coreA;
    ush*       e_out = (step & 1) ? e_coreA : e_coreB;
    const ush* x_in = (step & 1) ? x_coreB : x_coreA;
    ush*       x_out = (step & 1) ? x_coreA : x_coreB;
    const ush* u_in = (step & 1) ? u_coreB : u_coreA;
    ush*       u_out = (step & 1) ? u_coreA : u_coreB;

    hipMemsetAsync(agg, 0, (size_t)N * 64 * 4, stream);
    hipMemsetAsync(e_sum, 0, (size_t)G * 64 * 4, stream);
    hipMemsetAsync(n_sum, 0, (size_t)G * 64 * 4, stream);

    edge_mlp_kernel<<<edgeBlocks, 128, 0, stream>>>(
        ei0, ei1, batch, e_enc_bf, x_enc_bf, u_enc_bf, e_in, x_in, u_in, W1Te,
        em_l1_b, W2Te, em_l2_b, em_ln_g, em_ln_b, e_out, agg, e_sum, E2, E);

    f32_to_bf16_kernel<<<(N * 64 + 255) / 256, 256, 0, stream>>>(agg, agg_bf,
                                                                 N * 64);

    node_mlp_kernel<<<nodeBlocks, 128, 0, stream>>>(
        batch, x_enc_bf, u_enc_bf, x_in, agg_bf, u_in, W1Tn, nm_l1_b, W2Tn,
        nm_l2_b, nm_ln_g, nm_ln_b, vw, x_core_f32, x_out, n_sum, N);

    glob_step_kernel<<<1, 64, 0, stream>>>(u_enc_bf, u_in, e_sum, n_sum, ecnt,
                                           vcnt, gm_l1_w, gm_l1_b, gm_l2_w,
                                           gm_l2_b, gm_ln_g, gm_ln_b, u_out);
  }

  decode_kernel<<<(N + 255) / 256, 256, 0, stream>>>(
      x_core_f32, nd_lin_w, nd_lin_b, nd_ln_g, nd_ln_b, q_w, q_b, out_q, N);
}
